// GraphTransformer_49993419325531
// MI455X (gfx1250) — matrix operand order repaired and checked by reading, not compiled, not run on hardware
//
#include <hip/hip_runtime.h>
#include <cstdint>

// Graphormer forward for MI455X (gfx1250): all matmuls via v_wmma_f32_16x16x32_f16.
// Attention fused (scores+softmax+PV) per 16-query tile in LDS. Spatial bias is
// precomputed once in WMMA C-fragment layout and fed as the WMMA accumulator;
// the 1/sqrt(dk) scale is folded into the q projection; softmax normalization is
// folded into the PV accumulators.

typedef _Float16 f16;
typedef f16   v16h __attribute__((ext_vector_type(16)));
typedef float v8f  __attribute__((ext_vector_type(8)));

#define DEV __device__ __forceinline__

constexpr int Bc = 4, Nc = 1024, Hc = 8, Dc = 256, Lc = 4, DKc = 32;
constexpr int MROWS = Bc * Nc;            // 4096 flattened rows
constexpr float EPSc = 1e-6f;
constexpr float SCALEc = 0.17677669529663687f; // 1/sqrt(32)

// ---------------- WMMA fragment helpers (ISA 7.12.2 layouts) ----------------

// A-matrix 16x32 f16, source row-major with leading dim ld (elements).
// lane<16: row=lane, K=0..7 (VGPR0-3) and 16..23 (VGPR4-7)
// lane>=16: row=lane-16, K=8..15 and 24..31  -> two b128 loads per lane.
DEV v16h load_a16x32(const f16* base, int ld) {
  int lane = threadIdx.x & 31;
  int row  = lane & 15;
  int kb   = (lane >> 4) * 8;
  union { v16h v; uint4 q[2]; } a;
  a.q[0] = *reinterpret_cast<const uint4*>(base + row * ld + kb);
  a.q[1] = *reinterpret_cast<const uint4*>(base + row * ld + 16 + kb);
  return a.v;
}

// B-matrix 32x16 f16 from a source stored [n][k] (K contiguous, stride ld).
// lane<16: col=lane, K=0..7 and 16..23 ; lane>=16: col=lane-16, K=8..15 and 24..31.
DEV v16h load_b32x16(const f16* base, int ld) {
  int lane = threadIdx.x & 31;
  int col  = lane & 15;
  int kb   = (lane >> 4) * 8;
  union { v16h v; uint4 q[2]; } a;
  const f16* p = base + col * ld + kb;
  a.q[0] = *reinterpret_cast<const uint4*>(p);
  a.q[1] = *reinterpret_cast<const uint4*>(p + 16);
  return a.v;
}

DEV v8f wmma_f16(v16h a, v16h b, v8f c) {
  return __builtin_amdgcn_wmma_f32_16x16x32_f16(false, a, false, b, (short)0, c,
                                                false, false);
}

// ---------------- weight convert+transpose: W[l][k][n] f32 -> Wt[l][n][k] f16 ----

__global__ __launch_bounds__(256) void convw_kernel(const float* __restrict__ src,
                                                    f16* __restrict__ dst) {
  int idx = blockIdx.x * 256 + threadIdx.x;       // over Lc*Dc*Dc = 262144
  int kk = idx & 255;
  int n  = (idx >> 8) & 255;
  int l  = idx >> 16;
  dst[((size_t)l << 16) + (n << 8) + kk] = (f16)src[((size_t)l << 16) + (kk << 8) + n];
}

// ------- spatial bias precompute in C-fragment layout (layer-invariant) ----------
// biasF[h][qt][kt][lane][r] : value for score row m = (lane>>4)*8+r of q-tile qt,
// key = kt*16 + (lane&15). Each lane's 8 values are contiguous -> one b128 load.

__global__ __launch_bounds__(256) void bias_kernel(const int* __restrict__ spos,
                                                   const float* __restrict__ spemb,
                                                   f16* __restrict__ biasF) {
  int idx = blockIdx.x * 256 + threadIdx.x;       // over H*64*64*32*8 = 8M
  int r    = idx & 7;
  int lane = (idx >> 3) & 31;
  int kt   = (idx >> 8) & 63;
  int qt   = (idx >> 14) & 63;
  int h    = idx >> 20;
  int qrow = qt * 16 + (lane >> 4) * 8 + r;
  int key  = kt * 16 + (lane & 15);
  biasF[idx] = (f16)spemb[spos[qrow * Nc + key] * Hc + h];
}

// ---------------- embedding prologue: h = x + degEmb + svd@W + b ----------------

__global__ __launch_bounds__(256) void embed_kernel(
    const float* __restrict__ x, const int* __restrict__ indeg,
    const int* __restrict__ outdeg, const float* __restrict__ svde,
    const float* __restrict__ inde, const float* __restrict__ outde,
    const float* __restrict__ Wsvd, const float* __restrict__ bsvd,
    float* __restrict__ hf, f16* __restrict__ hh) {
  int idx = blockIdx.x * 256 + threadIdx.x;       // over B*N*D
  int d = idx & 255;
  int n = (idx >> 8) & 1023;
  float acc = x[idx] + inde[indeg[n] * Dc + d] + outde[outdeg[n] * Dc + d] + bsvd[d];
  const float* sv = svde + n * 32;
#pragma unroll
  for (int j = 0; j < 16; ++j) acc += sv[j] * Wsvd[j * Dc + d];
#pragma unroll
  for (int j = 16; j < 32; ++j) acc -= sv[j] * Wsvd[j * Dc + d];
  hf[idx] = acc;
  hh[idx] = (f16)acc;
}

// ---------------- generic WMMA GEMM with LDS-staged weight tile ------------------
// Y[4096,256] = X f16 @ Wt^T + bias. Block = 4 waves, each wave a 16x64 tile;
// the shared 64x256 weight tile (32KB, contiguous in Wt) is staged in LDS once.

enum { MODE_F16 = 0, MODE_RELU = 1, MODE_VT = 2, MODE_RES = 3, MODE_QSC = 4 };

template <int MODE>
__global__ __launch_bounds__(128) void gemm_kernel(
    const f16* __restrict__ X, const f16* __restrict__ Wt,
    const float* __restrict__ bias, f16* __restrict__ outh,
    float* __restrict__ outf, const float* __restrict__ res) {
  __shared__ f16 sW[64 * Dc];                     // 32 KB: rows n0..n0+63 of Wt
  int n0 = blockIdx.y * 64;
  {
    const uint4* src = reinterpret_cast<const uint4*>(Wt + (size_t)n0 * Dc);
    uint4* dst = reinterpret_cast<uint4*>(sW);
#pragma unroll
    for (int i = 0; i < 16; ++i)
      dst[threadIdx.x + i * 128] = src[threadIdx.x + i * 128];
  }
  __syncthreads();

  int wave = threadIdx.x >> 5;
  int m0 = (blockIdx.x * 4 + wave) * 16;
  v8f acc[4];
#pragma unroll
  for (int t = 0; t < 4; ++t) acc[t] = v8f{};
#pragma unroll
  for (int kt = 0; kt < Dc / 32; ++kt) {
    v16h a = load_a16x32(X + (size_t)m0 * Dc + kt * 32, Dc);
#pragma unroll
    for (int t = 0; t < 4; ++t) {
      v16h b = load_b32x16(sW + (t * 16) * Dc + kt * 32, Dc);
      acc[t] = wmma_f16(a, b, acc[t]);
    }
  }
  int lane = threadIdx.x & 31;
  int col = lane & 15, rbase = (lane >> 4) * 8;
#pragma unroll
  for (int t = 0; t < 4; ++t) {
    int n = n0 + t * 16 + col;
    float bv = bias[n];
#pragma unroll
    for (int r = 0; r < 8; ++r) {
      int m = m0 + rbase + r;
      float y = acc[t][r] + bv;
      size_t idx = (size_t)m * Dc + n;
      if (MODE == MODE_F16) {
        outh[idx] = (f16)y;
      } else if (MODE == MODE_QSC) {  // q projection: fold in 1/sqrt(dk)
        outh[idx] = (f16)(y * SCALEc);
      } else if (MODE == MODE_RELU) {
        outh[idx] = (f16)fmaxf(y, 0.0f);
      } else if (MODE == MODE_VT) {   // vT[b][n][m%N] for contiguous PV B-fragments
        int bb = m >> 10, nn = m & 1023;
        outh[((size_t)bb * Dc + n) * Nc + nn] = (f16)y;
      } else {                        // MODE_RES: pre-LN f32 = y + residual
        outf[idx] = y + res[idx];
      }
    }
  }
}

// ---------------- fused attention: scores -> softmax -> P@V, per 16-query tile ----

__global__ __launch_bounds__(32) void attn_kernel(
    const f16* __restrict__ q, const f16* __restrict__ k,
    const f16* __restrict__ vT, const f16* __restrict__ biasF,
    f16* __restrict__ o) {
  __shared__ f16 sP[16 * Nc];                     // 32 KB: 16 rows x 1024 keys
  int bh = blockIdx.y;
  int b = bh >> 3, h = bh & 7;
  int qt = blockIdx.x;
  int lane = threadIdx.x;
  int col = lane & 15, rbase = (lane >> 4) * 8;

  // ---- pass 1: S = q_scaled @ k^T + bias (bias enters as the WMMA C operand)
  const f16* qbase = q + ((size_t)b * Nc + qt * 16) * Dc + h * DKc;
  const f16* bfr   = biasF + (((size_t)h * 64 + qt) * 64) * 256 + lane * 8;
  v16h afrag = load_a16x32(qbase, Dc);            // 16 queries x 32 dk (pre-scaled)
  union H8 { uint4 u; f16 h[8]; };
#pragma unroll 4
  for (int kt = 0; kt < Nc / 16; ++kt) {
    const f16* kbase = k + ((size_t)b * Nc + kt * 16) * Dc + h * DKc;
    v16h bfrag = load_b32x16(kbase, Dc);          // keys are columns, dk contiguous
    H8 bt;
    bt.u = *reinterpret_cast<const uint4*>(bfr + kt * 256);
    v8f c;
#pragma unroll
    for (int e = 0; e < 8; ++e) c[e] = (float)bt.h[e];
    c = wmma_f16(afrag, bfrag, c);                // D = q*k^T + bias
    int key = kt * 16 + col;
#pragma unroll
    for (int r = 0; r < 8; ++r)
      sP[(rbase + r) * Nc + key] = (f16)c[r];
  }
  __syncthreads();

  // ---- pass 2: row max + exp (unnormalized) in place, vectorized by 8 halves
  int row = lane & 15;
  uint4* prow = reinterpret_cast<uint4*>(sP + row * Nc + (lane >> 4) * 512);
  float mx = -1e30f;
  for (int j = 0; j < 64; ++j) {
    H8 t; t.u = prow[j];
#pragma unroll
    for (int e = 0; e < 8; ++e) mx = fmaxf(mx, (float)t.h[e]);
  }
  mx = fmaxf(mx, __shfl_xor(mx, 16));
  float s = 0.0f;
  for (int j = 0; j < 64; ++j) {
    H8 t; t.u = prow[j];
#pragma unroll
    for (int e = 0; e < 8; ++e) {
      float ev = __expf((float)t.h[e] - mx);
      s += ev;
      t.h[e] = (f16)ev;
    }
    prow[j] = t.u;
  }
  s += __shfl_xor(s, 16);
  float inv = 1.0f / s;                           // lane L holds inv for row L&15
  __syncthreads();

  // ---- pass 3: O = (P_unnorm @ V) * inv, A-fragments from LDS, B from vT
  v8f acc0 = v8f{}, acc1 = v8f{};
  const f16* vbase = vT + ((size_t)b * Dc + h * DKc) * Nc;  // [dk][key]
#pragma unroll 4
  for (int kt = 0; kt < Nc / 32; ++kt) {
    v16h ap = load_a16x32(sP + kt * 32, Nc);
    v16h b0 = load_b32x16(vbase + kt * 32, Nc);             // dk cols 0..15
    v16h b1 = load_b32x16(vbase + 16 * Nc + kt * 32, Nc);   // dk cols 16..31
    acc0 = wmma_f16(ap, b0, acc0);
    acc1 = wmma_f16(ap, b1, acc1);
  }
  f16* obase = o + ((size_t)b * Nc + qt * 16) * Dc + h * DKc;
#pragma unroll
  for (int r = 0; r < 8; ++r) {
    int m = rbase + r;
    float invr = __shfl(inv, m);                  // inv for row m lives in lane m
    obase[m * Dc + col]      = (f16)(acc0[r] * invr);
    obase[m * Dc + col + 16] = (f16)(acc1[r] * invr);
  }
}

// ---------------- LayerNorm: one wave per row; optional extra f32 output ----------

__global__ __launch_bounds__(32) void ln_kernel(
    const float* __restrict__ xin, const float* __restrict__ g,
    const float* __restrict__ bta, float* __restrict__ hf,
    f16* __restrict__ hh, float* __restrict__ extra) {
  int row = blockIdx.x, lane = threadIdx.x;
  const float* xr = xin + (size_t)row * Dc;
  float vals[8];
  float s = 0.0f;
#pragma unroll
  for (int j = 0; j < 8; ++j) { vals[j] = xr[lane * 8 + j]; s += vals[j]; }
#pragma unroll
  for (int off = 16; off > 0; off >>= 1) s += __shfl_xor(s, off);
  float mu = s * (1.0f / Dc);
  float v = 0.0f;
#pragma unroll
  for (int j = 0; j < 8; ++j) { float d = vals[j] - mu; v += d * d; }
#pragma unroll
  for (int off = 16; off > 0; off >>= 1) v += __shfl_xor(v, off);
  float rinv = rsqrtf(v * (1.0f / Dc) + EPSc);
#pragma unroll
  for (int j = 0; j < 8; ++j) {
    int n = lane * 8 + j;
    float y = (vals[j] - mu) * rinv * g[n] + bta[n];
    size_t idx = (size_t)row * Dc + n;
    hf[idx] = y;
    hh[idx] = (f16)y;
    if (extra) extra[idx] = y;
  }
}

// ---------------- host orchestration ----------------

extern "C" void kernel_launch(void* const* d_in, const int* in_sizes, int n_in,
                              void* d_out, int out_size, void* d_ws, size_t ws_size,
                              hipStream_t stream) {
  (void)in_sizes; (void)n_in; (void)out_size; (void)ws_size;
  const float* x      = (const float*)d_in[0];
  const int*   indeg  = (const int*)d_in[1];
  const int*   outdeg = (const int*)d_in[2];
  const int*   spos   = (const int*)d_in[3];
  const float* svde   = (const float*)d_in[4];
  const float* inde   = (const float*)d_in[5];
  const float* outde  = (const float*)d_in[6];
  const float* spemb  = (const float*)d_in[7];
  const float* Wsvd   = (const float*)d_in[8];
  const float* bsvd   = (const float*)d_in[9];
  const float* Wsrc[6] = {(const float*)d_in[10], (const float*)d_in[11],
                          (const float*)d_in[12], (const float*)d_in[13],
                          (const float*)d_in[14], (const float*)d_in[15]};
  const float* bvec[6] = {(const float*)d_in[16], (const float*)d_in[17],
                          (const float*)d_in[18], (const float*)d_in[19],
                          (const float*)d_in[20], (const float*)d_in[21]};
  const float* ln1b = (const float*)d_in[22];
  const float* ln2b = (const float*)d_in[23];
  const float* ln1g = (const float*)d_in[24];
  const float* ln2g = (const float*)d_in[25];

  uint8_t* wsp = (uint8_t*)d_ws;
  float* hf    = (float*)(wsp);                      // 4 MB  fp32 activations
  f16*   hh    = (f16*)(wsp + (4u << 20));           // 2 MB  fp16 activations
  float* tmpf  = (float*)(wsp + (6u << 20));         // 4 MB  pre-LN fp32
  f16*   qb    = (f16*)(wsp + (10u << 20));          // 2 MB  (pre-scaled by 1/sqrt(dk))
  f16*   kb    = (f16*)(wsp + (12u << 20));          // 2 MB
  f16*   vT    = (f16*)(wsp + (14u << 20));          // 2 MB  v transposed [B][D][N]
  f16*   ob    = (f16*)(wsp + (16u << 20));          // 2 MB  attention out
  f16*   ub    = (f16*)(wsp + (18u << 20));          // 2 MB  FFN hidden
  f16*   Wt    = (f16*)(wsp + (20u << 20));          // 3 MB  all weights, f16 [n][k]
  f16*   biasF = (f16*)(wsp + (23u << 20));          // 16 MB bias f16, C-frag layout

  const size_t WSZ = (size_t)Lc * Dc * Dc;           // per weight-set stride (elems)
  for (int wdx = 0; wdx < 6; ++wdx)
    convw_kernel<<<1024, 256, 0, stream>>>(Wsrc[wdx], Wt + wdx * WSZ);

  bias_kernel<<<(Hc * Nc * Nc) / 256, 256, 0, stream>>>(spos, spemb, biasF);

  embed_kernel<<<4096, 256, 0, stream>>>(x, indeg, outdeg, svde, inde, outde,
                                         Wsvd, bsvd, hf, hh);

  dim3 ggrid(MROWS / 64, Dc / 64);                   // (64, 4)
  dim3 gblk(128);
  dim3 agrid(Nc / 16, Bc * Hc);                      // (64, 32)

  for (int l = 0; l < Lc; ++l) {
    const f16* Wq = Wt + 0 * WSZ + (size_t)l * Dc * Dc;
    const f16* Wk = Wt + 1 * WSZ + (size_t)l * Dc * Dc;
    const f16* Wv = Wt + 2 * WSZ + (size_t)l * Dc * Dc;
    const f16* Wa = Wt + 3 * WSZ + (size_t)l * Dc * Dc;
    const f16* W1 = Wt + 4 * WSZ + (size_t)l * Dc * Dc;
    const f16* W2 = Wt + 5 * WSZ + (size_t)l * Dc * Dc;

    gemm_kernel<MODE_QSC><<<ggrid, gblk, 0, stream>>>(hh, Wq, bvec[0] + l * Dc, qb, nullptr, nullptr);
    gemm_kernel<MODE_F16><<<ggrid, gblk, 0, stream>>>(hh, Wk, bvec[1] + l * Dc, kb, nullptr, nullptr);
    gemm_kernel<MODE_VT ><<<ggrid, gblk, 0, stream>>>(hh, Wv, bvec[2] + l * Dc, vT, nullptr, nullptr);

    attn_kernel<<<agrid, 32, 0, stream>>>(qb, kb, vT, biasF, ob);

    gemm_kernel<MODE_RES><<<ggrid, gblk, 0, stream>>>(ob, Wa, bvec[3] + l * Dc, nullptr, tmpf, hf);
    ln_kernel<<<MROWS, 32, 0, stream>>>(tmpf, ln1g + l * Dc, ln1b + l * Dc, hf, hh, nullptr);

    gemm_kernel<MODE_RELU><<<ggrid, gblk, 0, stream>>>(hh, W1, bvec[4] + l * Dc, ub, nullptr, nullptr);
    gemm_kernel<MODE_RES ><<<ggrid, gblk, 0, stream>>>(ub, W2, bvec[5] + l * Dc, nullptr, tmpf, hf);
    ln_kernel<<<MROWS, 32, 0, stream>>>(tmpf, ln2g + l * Dc, ln2b + l * Dc, hf, hh,
                                        (l == Lc - 1) ? (float*)d_out : nullptr);
  }
}
